// TransformerBlock_29661044146183
// MI455X (gfx1250) — matrix operand order repaired and checked by reading, not compiled, not run on hardware
//
#include <hip/hip_runtime.h>
#include <math.h>

typedef __bf16 bf16_t;
typedef __attribute__((ext_vector_type(16))) __bf16 bf16x16;
typedef __attribute__((ext_vector_type(8)))  float  floatx8;
typedef unsigned int __attribute__((ext_vector_type(4))) uint32x4;
typedef int  __attribute__((ext_vector_type(8))) int32x8;
typedef int  __attribute__((ext_vector_type(4))) int32x4;

#define EMBED      1024
#define NHEAD      16
#define HDIM       64
#define HIDDEN_DIM 4096
#define BATCH      4
#define SEQ        1024
#define ROWS       (BATCH * SEQ)   // 4096

// LDS tile geometry: 128 rows x 32 bf16, row stride 40 elems (64B data + 16B pad)
#define TILE_K      32
#define TILE_ROWS   128
#define LDS_STRIDE  40

#if defined(__gfx1250__) && __has_builtin(__builtin_amdgcn_tensor_load_to_lds) && \
    __has_builtin(__builtin_amdgcn_s_wait_tensorcnt)
#define USE_TDM 1
#else
#define USE_TDM 0
#endif

// ------------------------------------------------------------------
// W [K,N] f32 row-major -> Wt [N,K] bf16 row-major (transposed cast)
// ------------------------------------------------------------------
__global__ void k_convert_transpose(const float* __restrict__ W,
                                    bf16_t* __restrict__ Wt,
                                    int K, int N) {
  long i = (long)blockIdx.x * blockDim.x + threadIdx.x;
  long total = (long)K * N;
  if (i >= total) return;
  int n = (int)(i % N);
  int k = (int)(i / N);
  Wt[(long)n * K + k] = (bf16_t)W[i];
}

// ------------------------------------------------------------------
// LayerNorm over C=1024, one block (256 thr) per row, bf16 output
// ------------------------------------------------------------------
__global__ void k_layernorm(const float* __restrict__ x,
                            const float* __restrict__ g,
                            const float* __restrict__ b,
                            bf16_t* __restrict__ out) {
  const int C = EMBED;
  int row = blockIdx.x;
  const float* xr = x + (long)row * C;
  bf16_t* orow = out + (long)row * C;

  float v[4], s = 0.f, ss = 0.f;
#pragma unroll
  for (int i = 0; i < 4; ++i) {
    float t = xr[threadIdx.x + i * 256];
    v[i] = t; s += t; ss += t * t;
  }
  __shared__ float r1[256], r2[256];
  r1[threadIdx.x] = s; r2[threadIdx.x] = ss;
  __syncthreads();
  for (int off = 128; off > 0; off >>= 1) {
    if (threadIdx.x < off) {
      r1[threadIdx.x] += r1[threadIdx.x + off];
      r2[threadIdx.x] += r2[threadIdx.x + off];
    }
    __syncthreads();
  }
  float mu   = r1[0] * (1.f / C);
  float var  = r2[0] * (1.f / C) - mu * mu;
  float rstd = rsqrtf(var + 1e-5f);
#pragma unroll
  for (int i = 0; i < 4; ++i) {
    int c = threadIdx.x + i * 256;
    orow[c] = (bf16_t)((v[i] - mu) * rstd * g[c] + b[c]);
  }
}

// ------------------------------------------------------------------
// Row softmax over 1024 cols: S fp32 -> P bf16 (rows flat across b,h)
// ------------------------------------------------------------------
__global__ void k_softmax(const float* __restrict__ S,
                          bf16_t* __restrict__ P) {
  const int N = SEQ;
  long row = blockIdx.x;
  const float* sr = S + row * N;
  bf16_t* pr = P + row * N;

  float v[4], m = -1e30f;
#pragma unroll
  for (int i = 0; i < 4; ++i) {
    v[i] = sr[threadIdx.x + i * 256];
    m = fmaxf(m, v[i]);
  }
  __shared__ float red[256];
  red[threadIdx.x] = m; __syncthreads();
  for (int off = 128; off > 0; off >>= 1) {
    if (threadIdx.x < off)
      red[threadIdx.x] = fmaxf(red[threadIdx.x], red[threadIdx.x + off]);
    __syncthreads();
  }
  m = red[0];
  __syncthreads();
  float s = 0.f;
#pragma unroll
  for (int i = 0; i < 4; ++i) { v[i] = __expf(v[i] - m); s += v[i]; }
  red[threadIdx.x] = s; __syncthreads();
  for (int off = 128; off > 0; off >>= 1) {
    if (threadIdx.x < off) red[threadIdx.x] += red[threadIdx.x + off];
    __syncthreads();
  }
  float inv = 1.f / red[0];
#pragma unroll
  for (int i = 0; i < 4; ++i)
    pr[threadIdx.x + i * 256] = (bf16_t)(v[i] * inv);
}

// ------------------------------------------------------------------
// V slice of qkv -> Vt[b][h][d][n]  (so PV GEMM's B operand is contiguous)
// ------------------------------------------------------------------
__global__ void k_transpose_v(const bf16_t* __restrict__ qkv,
                              bf16_t* __restrict__ vt) {
  long i = (long)blockIdx.x * blockDim.x + threadIdx.x;
  long total = (long)BATCH * NHEAD * HDIM * SEQ;
  if (i >= total) return;
  int n = (int)(i % SEQ);
  long r = i / SEQ;
  int d = (int)(r % HDIM); r /= HDIM;
  int h = (int)(r % NHEAD);
  int b = (int)(r / NHEAD);
  vt[i] = qkv[((long)(b * SEQ + n)) * (3 * EMBED) + 2 * EMBED + h * HDIM + d];
}

// ------------------------------------------------------------------
// Tile staging: global [rows x 32] (row stride ld elems) -> LDS
// (row stride LDS_STRIDE elems, 16B pad per row for bank spread)
// ------------------------------------------------------------------
#if USE_TDM
union G1U { unsigned u[8]; int32x8 v; };

__device__ __forceinline__ void tdm_issue(const bf16_t* g, int ldElems,
                                          int rowsValid, bf16_t* lds) {
  unsigned ldsAddr = (unsigned)(size_t)lds;            // low 32 bits = LDS offset
  unsigned long long ga = (unsigned long long)(size_t)g;
  uint32x4 g0;
  g0.x = 1u;                                           // count=1, user descriptor
  g0.y = ldsAddr;                                      // lds_addr
  g0.z = (unsigned)ga;                                 // global_addr[31:0]
  g0.w = ((unsigned)(ga >> 32) & 0x01FFFFFFu) | (2u << 30);  // addr[56:32], type=2

  unsigned tdim0 = TILE_K;                 // full K chunk always valid
  unsigned tdim1 = (unsigned)rowsValid;    // rows beyond -> zero-filled by TDM
  unsigned long long s0 = (unsigned long long)ldElems; // dim0 stride (elems)
  G1U g1;
  // data_size=1 (2B) @16 | pad_enable @20 | pad_interval=3 (16 dw = 64B) @22
  // | pad_amount=3 (4 dw = 16B) @25
  g1.u[0] = (1u << 16) | (1u << 20) | (3u << 22) | (3u << 25);
  g1.u[1] = (tdim0 & 0xFFFFu) << 16;                       // tensor_dim0 lo
  g1.u[2] = (tdim0 >> 16) | ((tdim1 & 0xFFFFu) << 16);     // dim0 hi | dim1 lo
  g1.u[3] = (tdim1 >> 16) | ((unsigned)TILE_K << 16);      // dim1 hi | tile_dim0
  g1.u[4] = (unsigned)TILE_ROWS;                           // tile_dim1 | tile_dim2=0
  g1.u[5] = (unsigned)(s0 & 0xFFFFFFFFu);                  // dim0_stride lo
  g1.u[6] = (unsigned)((s0 >> 32) & 0xFFFFu);              // dim0_stride hi | dim1_stride lo
  g1.u[7] = 0u;                                            // dim1_stride hi
  int32x4 z4 = {0, 0, 0, 0};
  int32x8 z8 = {0, 0, 0, 0, 0, 0, 0, 0};
  // 6-arg variant (clang-23 / therock-10.0 headers): groups 2/3 + extra group zeroed
  __builtin_amdgcn_tensor_load_to_lds(g0, g1.v, z4, z4, z8, 0);
}
#else
__device__ __forceinline__ void coop_copy_tile(const bf16_t* __restrict__ g,
                                               int ld, int rowsValid,
                                               bf16_t* __restrict__ s, int tid) {
#pragma unroll
  for (int i = 0; i < 2; ++i) {
    int e = (tid + i * 256) * 8;     // 8 bf16 = 16B per chunk
    int r = e >> 5;
    int kk = e & 31;
    int rr = r < rowsValid ? r : (rowsValid - 1);
    *(uint4*)(s + r * LDS_STRIDE + kk) = *(const uint4*)(g + (size_t)rr * ld + kk);
  }
}
#endif

// ------------------------------------------------------------------
// WMMA fragment loaders from LDS tile (layouts per CDNA5 ISA 7.12.2)
// ------------------------------------------------------------------
__device__ __forceinline__ bf16x16 lds_a_frag(const bf16_t* tile, int row, int lane) {
  int r  = row + (lane & 15);
  int kb = (lane >> 4) << 3;               // lanes 0-15: +0, 16-31: +8
  const bf16_t* p = tile + r * LDS_STRIDE + kb;
  bf16x16 f;
  ((uint4*)&f)[0] = *(const uint4*)(p);      // K kb..kb+7
  ((uint4*)&f)[1] = *(const uint4*)(p + 16); // K kb+16..kb+23
  return f;
}
__device__ __forceinline__ bf16x16 lds_b_frag(const bf16_t* tile, int col, int lane) {
  int c  = col + (lane & 15);
  int kb = (lane >> 4) << 3;               // lanes 0-15: +0, 16-31: +8
  const bf16_t* p = tile + c * LDS_STRIDE + kb;
  bf16x16 f;
  ((uint4*)&f)[0] = *(const uint4*)(p);      // K kb..kb+7
  ((uint4*)&f)[1] = *(const uint4*)(p + 16); // K kb+16..kb+23
  return f;
}

// ------------------------------------------------------------------
// LDS-staged, TDM-fed, double-buffered bf16 WMMA GEMM
//   out[z] = A[z][M,K] @ Bt[z][N,K]^T  (+fused epilogue)
// block = 256 thr = 8 waves (2x4); wave tile 64x32 = 4x2 WMMAs
// batch z decomposed as (b = z>>4, h = z&15) with stride pairs.
// MODE: 0 store bf16 | 1 +bias->bf16 | 2 +bias,gelu->bf16
//       3 +bias,+f32 residual->f32  | 4 *scale->f32
// ------------------------------------------------------------------
template <int MODE>
__global__ void __launch_bounds__(256)
k_gemm_wmma(const bf16_t* __restrict__ A, int lda, long sAb, long sAh,
            const bf16_t* __restrict__ Bt, int ldb, long sBb, long sBh,
            void* __restrict__ out, int ldc, long sCb, long sCh,
            int M, int N, int K,
            const float* __restrict__ bias,
            const float* __restrict__ resid,
            float scale) {
  __shared__ bf16_t Asm[2][TILE_ROWS * LDS_STRIDE];
  __shared__ bf16_t Bsm[2][TILE_ROWS * LDS_STRIDE];

  const int tid   = threadIdx.x;
  const int lane  = tid & 31;
  const int wid   = tid >> 5;
  const int waveM = wid >> 2;                   // 0..1
  const int waveN = wid & 3;                    // 0..3
  const int blkRow = blockIdx.y * 128;
  const int blkCol = blockIdx.x * 128;

  const int zz = blockIdx.z;
  const int hb = zz & (NHEAD - 1);
  const int bb = zz >> 4;
  A   += (size_t)bb * sAb + (size_t)hb * sAh;
  Bt  += (size_t)bb * sBb + (size_t)hb * sBh;
  char* outp = (char*)out;
  size_t outOff = (size_t)bb * sCb + (size_t)hb * sCh;

  const int aRows = M - blkRow;                 // always >= 128 in our launches
  const int bRows = N - blkCol;                 // may be < 128 (PV tail: zero-fill)
  const bf16_t* Ag = A  + (size_t)blkRow * lda;
  const bf16_t* Bg = Bt + (size_t)blkCol * ldb;

  floatx8 acc[4][2] = {};

  // ---- prologue: stage k=0 into buffer 0 ----
#if USE_TDM
  if (tid < 32) {
    tdm_issue(Ag, lda, aRows, Asm[0]);
    tdm_issue(Bg, ldb, bRows, Bsm[0]);
  }
  __builtin_amdgcn_s_wait_tensorcnt(0);
#else
  coop_copy_tile(Ag, lda, aRows, Asm[0], tid);
  coop_copy_tile(Bg, ldb, bRows, Bsm[0], tid);
#endif
  __syncthreads();

  int buf = 0;
  for (int k = 0; k < K; k += TILE_K) {
    int nk = k + TILE_K;
    if (nk < K) {   // stage next tile into the other buffer (overlaps compute)
#if USE_TDM
      if (tid < 32) {
        tdm_issue(Ag + nk, lda, aRows, Asm[buf ^ 1]);
        tdm_issue(Bg + nk, ldb, bRows, Bsm[buf ^ 1]);
      }
#else
      coop_copy_tile(Ag + nk, lda, aRows, Asm[buf ^ 1], tid);
      coop_copy_tile(Bg + nk, ldb, bRows, Bsm[buf ^ 1], tid);
#endif
    }

    // ---- consume current buffer ----
    const bf16_t* As = Asm[buf];
    const bf16_t* Bs = Bsm[buf];
    bf16x16 af[4], bfr[2];
#pragma unroll
    for (int mi = 0; mi < 4; ++mi)
      af[mi] = lds_a_frag(As, waveM * 64 + mi * 16, lane);
#pragma unroll
    for (int ni = 0; ni < 2; ++ni)
      bfr[ni] = lds_b_frag(Bs, waveN * 32 + ni * 16, lane);
#pragma unroll
    for (int mi = 0; mi < 4; ++mi)
#pragma unroll
      for (int ni = 0; ni < 2; ++ni)
        acc[mi][ni] = __builtin_amdgcn_wmma_f32_16x16x32_bf16(
            false, af[mi], false, bfr[ni], (short)0, acc[mi][ni], false, false);

#if USE_TDM
    __builtin_amdgcn_s_wait_tensorcnt(0);
#endif
    __syncthreads();
    buf ^= 1;
  }

  // ---- fused epilogue ----
#pragma unroll
  for (int ni = 0; ni < 2; ++ni) {
    int col = blkCol + waveN * 32 + ni * 16 + (lane & 15);
    if (col >= N) continue;
    float bv = 0.f;
    if (MODE == 1 || MODE == 2 || MODE == 3) bv = bias[col];
#pragma unroll
    for (int mi = 0; mi < 4; ++mi) {
      int rowTop = blkRow + waveM * 64 + mi * 16 + ((lane >> 4) << 3);
#pragma unroll
      for (int v = 0; v < 8; ++v) {
        int row = rowTop + v;
        if (row >= M) continue;
        float val = acc[mi][ni][v];
        if (MODE == 1 || MODE == 2 || MODE == 3) val += bv;
        if (MODE == 2) {  // tanh GELU (jax.nn.gelu approximate)
          float t = 0.7978845608028654f * (val + 0.044715f * val * val * val);
          val = 0.5f * val * (1.0f + tanhf(t));
        }
        if (MODE == 4) val *= scale;
        size_t idx = outOff + (size_t)row * ldc + col;
        if (MODE == 3)      ((float*)outp)[idx]  = val + resid[idx];
        else if (MODE == 4) ((float*)outp)[idx]  = val;
        else                ((bf16_t*)outp)[idx] = (bf16_t)val;
      }
    }
  }
}

// ------------------------------------------------------------------
extern "C" void kernel_launch(void* const* d_in, const int* in_sizes, int n_in,
                              void* d_out, int out_size, void* d_ws, size_t ws_size,
                              hipStream_t stream) {
  const float* x      = (const float*)d_in[0];
  const float* ln1_g  = (const float*)d_in[1];
  const float* ln1_b  = (const float*)d_in[2];
  const float* qkv_w  = (const float*)d_in[3];
  const float* qkv_b  = (const float*)d_in[4];
  const float* proj_w = (const float*)d_in[5];
  const float* proj_b = (const float*)d_in[6];
  const float* ln2_g  = (const float*)d_in[7];
  const float* ln2_b  = (const float*)d_in[8];
  const float* fc1_w  = (const float*)d_in[9];
  const float* fc1_b  = (const float*)d_in[10];
  const float* fc2_w  = (const float*)d_in[11];
  const float* fc2_b  = (const float*)d_in[12];
  float* out = (float*)d_out;
  (void)in_sizes; (void)n_in; (void)out_size; (void)ws_size;

  char* wsp = (char*)d_ws;
  auto carve = [&](size_t bytes) {
    char* p = wsp;
    wsp += (bytes + 255) & ~(size_t)255;
    return (void*)p;
  };
  bf16_t* h1     = (bf16_t*)carve((size_t)ROWS * EMBED * 2);
  bf16_t* wqkvT  = (bf16_t*)carve((size_t)3 * EMBED * EMBED * 2);
  bf16_t* wprojT = (bf16_t*)carve((size_t)EMBED * EMBED * 2);
  bf16_t* wfc1T  = (bf16_t*)carve((size_t)HIDDEN_DIM * EMBED * 2);
  bf16_t* wfc2T  = (bf16_t*)carve((size_t)EMBED * HIDDEN_DIM * 2);
  bf16_t* qkv    = (bf16_t*)carve((size_t)ROWS * 3 * EMBED * 2);
  bf16_t* vt     = (bf16_t*)carve((size_t)BATCH * NHEAD * HDIM * SEQ * 2);
  float*  Sbuf   = (float*) carve((size_t)BATCH * NHEAD * SEQ * SEQ * 4);
  bf16_t* Pbuf   = (bf16_t*)carve((size_t)BATCH * NHEAD * SEQ * SEQ * 2);
  bf16_t* obuf   = (bf16_t*)carve((size_t)ROWS * EMBED * 2);
  float*  x1     = (float*) carve((size_t)ROWS * EMBED * 4);
  bf16_t* h2     = (bf16_t*)carve((size_t)ROWS * EMBED * 2);
  bf16_t* f1o    = (bf16_t*)carve((size_t)ROWS * HIDDEN_DIM * 2);

  dim3 blk(256);

  // ---- weight cast+transpose to bf16 ----
  { long t = (long)EMBED * 3 * EMBED;
    k_convert_transpose<<<(t + 255) / 256, blk, 0, stream>>>(qkv_w, wqkvT, EMBED, 3 * EMBED); }
  { long t = (long)EMBED * EMBED;
    k_convert_transpose<<<(t + 255) / 256, blk, 0, stream>>>(proj_w, wprojT, EMBED, EMBED); }
  { long t = (long)EMBED * HIDDEN_DIM;
    k_convert_transpose<<<(t + 255) / 256, blk, 0, stream>>>(fc1_w, wfc1T, EMBED, HIDDEN_DIM); }
  { long t = (long)HIDDEN_DIM * EMBED;
    k_convert_transpose<<<(t + 255) / 256, blk, 0, stream>>>(fc2_w, wfc2T, HIDDEN_DIM, EMBED); }

  // ---- attention branch ----
  k_layernorm<<<ROWS, blk, 0, stream>>>(x, ln1_g, ln1_b, h1);

  // qkv = h1 @ qkv_w + qkv_b  -> bf16 [4096, 3072]
  k_gemm_wmma<1><<<dim3(3 * EMBED / 128, ROWS / 128, 1), blk, 0, stream>>>(
      h1, EMBED, 0, 0, wqkvT, EMBED, 0, 0, qkv, 3 * EMBED, 0, 0,
      ROWS, 3 * EMBED, EMBED, qkv_b, nullptr, 1.f);

  { long t = (long)BATCH * NHEAD * HDIM * SEQ;
    k_transpose_v<<<(t + 255) / 256, blk, 0, stream>>>(qkv, vt); }

  // S[b,h] = (Q @ K^T) * 1/sqrt(D), all 64 heads batched over grid.z
  k_gemm_wmma<4><<<dim3(SEQ / 128, SEQ / 128, BATCH * NHEAD), blk, 0, stream>>>(
      qkv,         3 * EMBED, (long)SEQ * 3 * EMBED, HDIM,   // Q
      qkv + EMBED, 3 * EMBED, (long)SEQ * 3 * EMBED, HDIM,   // K (already Bt layout)
      Sbuf, SEQ, (long)NHEAD * SEQ * SEQ, (long)SEQ * SEQ,
      SEQ, SEQ, HDIM, nullptr, nullptr, 0.125f);

  k_softmax<<<BATCH * NHEAD * SEQ, blk, 0, stream>>>(Sbuf, Pbuf);

  // O[b,h] = P @ V  (N=64 tail handled by TDM zero-fill / clamped fallback)
  k_gemm_wmma<0><<<dim3(1, SEQ / 128, BATCH * NHEAD), blk, 0, stream>>>(
      Pbuf, SEQ, (long)NHEAD * SEQ * SEQ, (long)SEQ * SEQ,
      vt,   SEQ, (long)NHEAD * HDIM * SEQ, (long)HDIM * SEQ,
      obuf, EMBED, (long)SEQ * EMBED, (long)HDIM,
      SEQ, HDIM, SEQ, nullptr, nullptr, 1.f);

  // x1 = x + O @ proj_w + proj_b   (fp32)
  k_gemm_wmma<3><<<dim3(EMBED / 128, ROWS / 128, 1), blk, 0, stream>>>(
      obuf, EMBED, 0, 0, wprojT, EMBED, 0, 0, x1, EMBED, 0, 0,
      ROWS, EMBED, EMBED, proj_b, x, 1.f);

  // ---- MLP branch ----
  k_layernorm<<<ROWS, blk, 0, stream>>>(x1, ln2_g, ln2_b, h2);

  // f1o = gelu(h2 @ fc1_w + fc1_b)  -> bf16 [4096, 4096]
  k_gemm_wmma<2><<<dim3(HIDDEN_DIM / 128, ROWS / 128, 1), blk, 0, stream>>>(
      h2, EMBED, 0, 0, wfc1T, EMBED, 0, 0, f1o, HIDDEN_DIM, 0, 0,
      ROWS, HIDDEN_DIM, EMBED, fc1_b, nullptr, 1.f);

  // out = x1 + f1o @ fc2_w + fc2_b  (fp32)
  k_gemm_wmma<3><<<dim3(EMBED / 128, ROWS / 128, 1), blk, 0, stream>>>(
      f1o, HIDDEN_DIM, 0, 0, wfc2T, HIDDEN_DIM, 0, 0, out, EMBED, 0, 0,
      ROWS, EMBED, HIDDEN_DIM, fc2_b, x1, 1.f);
}
